// HyperFlowNet_37460704756484
// MI455X (gfx1250) — matrix operand order repaired and checked by reading, not compiled, not run on hardware
//
#include <hip/hip_runtime.h>
#include <hip/hip_bf16.h>
#include <math.h>

// ---------------------------------------------------------------------------
// HyperFlowNet (Transolver-style physics attention) for MI455X / gfx1250.
// wave32, WMMA f32_16x16x32_f16 for all large matmuls, fused FFN in LDS,
// async global->LDS staging (GLOBAL_LOAD_ASYNC_TO_LDS_B128 + s_wait_asynccnt).
// ---------------------------------------------------------------------------

typedef _Float16 f16;
typedef _Float16 h8   __attribute__((ext_vector_type(8)));
typedef _Float16 v16h __attribute__((ext_vector_type(16)));
typedef float    v8f  __attribute__((ext_vector_type(8)));

#define B_     4
#define N_     32768
#define BN_    (B_ * N_)        // 131072
#define WID_   256
#define DEPTH_ 8
#define M_     32
#define H_     8
#define DH_    32
#define FFN_   1024

// ---------------------------------------------------------------------------
// WMMA fragment helpers (layouts per CDNA5 ISA 7.12.2, wave32)
// A (16x32 f16, M x K): lanes 0-15 = rows; half h -> K = (h<8?0:16) + (lane>=16?8:0) + (h&7)
// B (32x16 f16, K x N): lane -> N = lane&15; half h -> K = (h<8?0:16) + (lane>=16?8:0) + (h&7)
// C/D (16x16 f32): lane -> N = lane&15; vgpr r -> M = r + (lane>=16?8:0)
// ---------------------------------------------------------------------------
__device__ __forceinline__ v16h load_a_frag(const f16* A, int lda, int lane) {
    const int m  = lane & 15;
    const int kb = (lane & 16) ? 8 : 0;
    const f16* row = A + (size_t)m * lda + kb;
    h8 lo = *(const h8*)row;         // K = kb .. kb+7
    h8 hi = *(const h8*)(row + 16);  // K = 16+kb .. 16+kb+7
    v16h a;
#pragma unroll
    for (int i = 0; i < 8; ++i) { a[i] = lo[i]; a[i + 8] = hi[i]; }
    return a;
}

// Wp is in "panel" layout: Wp[((k>>5)*N + n)*32 + (k&31)]
__device__ __forceinline__ v16h load_b_frag(const f16* Wp, int Nw, int nBase,
                                            int kt, int lane) {
    const int nn = lane & 15;
    const int kb = (lane & 16) ? 8 : 0;
    const f16* p = Wp + ((size_t)kt * Nw + nBase + nn) * 32 + kb;
    h8 lo = *(const h8*)p;
    h8 hi = *(const h8*)(p + 16);
    v16h b;
#pragma unroll
    for (int i = 0; i < 8; ++i) { b[i] = lo[i]; b[i + 8] = hi[i]; }
    return b;
}

__device__ __forceinline__ v8f wmma16(v16h a, v16h b, v8f c) {
    return __builtin_amdgcn_wmma_f32_16x16x32_f16(false, a, false, b,
                                                  (short)0, c, false, false);
}

__device__ __forceinline__ float gelu_exact(float v) {
    return 0.5f * v * (1.0f + erff(v * 0.70710678118654752f));
}

// Async 16B global -> LDS copy (CDNA5 GLOBAL_LOAD_ASYNC_TO_LDS_B128, ASYNCcnt).
// lds_off: byte offset within the workgroup LDS allocation (low 32 bits of the
// generic pointer on AMDGPU). Completion: async_wait_all() then __syncthreads().
__device__ __forceinline__ void async_ld_b128(unsigned lds_off, const void* gsrc) {
#if defined(__gfx1250__)
    asm volatile("global_load_async_to_lds_b128 %0, %1, off"
                 :: "v"(lds_off), "v"(gsrc) : "memory");
#else
    // host-pass / non-gfx1250 fallback (never executed on device)
    *(h8*)((char*)0 + lds_off) = *(const h8*)gsrc;
#endif
}
__device__ __forceinline__ void async_wait_all() {
#if defined(__gfx1250__)
    asm volatile("s_wait_asynccnt 0x0" ::: "memory");
#endif
}

// ---------------------------------------------------------------------------
// Weight pack: fp32 (K x N, row-major) -> f16 panel layout, K zero-padded to Kpad.
// blockIdx.y = layer.
// ---------------------------------------------------------------------------
__global__ void pack_w_kernel(const float* __restrict__ src, f16* __restrict__ dst,
                              int K, int Kpad, int N) {
    const size_t layer = blockIdx.y;
    src += layer * (size_t)K * N;
    dst += layer * (size_t)Kpad * N;
    int t = blockIdx.x * 256 + threadIdx.x;
    if (t >= Kpad * N) return;
    int k = t / N, n = t % N;
    float v = (k < K) ? src[(size_t)k * N + n] : 0.0f;
    dst[(((size_t)(k >> 5)) * N + n) * 32 + (k & 31)] = (f16)v;
}

// ---------------------------------------------------------------------------
// Encoding: [features(4) | sin(2pi c@F)(8) | cos(8) | sin(w*t*FB)(4) | cos(4) | pad(4)]
// -> f16 (BN x 32)
// ---------------------------------------------------------------------------
__global__ void encode_kernel(const float* __restrict__ feat,
                              const float* __restrict__ coords,
                              const float* __restrict__ t_norm,
                              const float* __restrict__ freq,   // (3,8)
                              const float* __restrict__ omega,  // (4)
                              f16* __restrict__ enc) {
    size_t t = (size_t)blockIdx.x * 256 + threadIdx.x;
    if (t >= BN_) return;
    int b = (int)(t / N_);
    float c0 = coords[t * 3 + 0], c1 = coords[t * 3 + 1], c2 = coords[t * 3 + 2];
    float e[32];
#pragma unroll
    for (int i = 0; i < 4; ++i) e[i] = feat[t * 4 + i];
#pragma unroll
    for (int j = 0; j < 8; ++j) {
        float p = 6.2831853071795864f *
                  (c0 * freq[j] + c1 * freq[8 + j] + c2 * freq[16 + j]);
        e[4 + j]  = sinf(p);
        e[12 + j] = cosf(p);
    }
    float tt = t_norm[b] * 1000.0f;
#pragma unroll
    for (int j = 0; j < 4; ++j) {
        float a = omega[j] * tt;
        e[20 + j] = sinf(a);
        e[24 + j] = cosf(a);
    }
#pragma unroll
    for (int j = 28; j < 32; ++j) e[j] = 0.0f;
    f16* d = enc + t * 32;
#pragma unroll
    for (int j = 0; j < 32; ++j) d[j] = (f16)e[j];
}

// ---------------------------------------------------------------------------
// Generic 1-wave WMMA GEMM: C(rows x Nw) f32 = A(rows x K) f16 @ Wp + bias
// grid.x = rows/16, grid.y = Nw/16, block = 32
// ---------------------------------------------------------------------------
__global__ __launch_bounds__(32) void wmma_gemm_kernel(
    const f16* __restrict__ A, int lda, const f16* __restrict__ Wp,
    const float* __restrict__ bias, float* __restrict__ C, int ldc,
    int K, int Nw) {
    const int lane = threadIdx.x & 31;
    const size_t rt = blockIdx.x;
    const int ct = blockIdx.y;
    v8f acc = {0.f, 0.f, 0.f, 0.f, 0.f, 0.f, 0.f, 0.f};
    const int nkt = K >> 5;
    for (int kt = 0; kt < nkt; ++kt) {
        v16h a = load_a_frag(A + rt * 16 * (size_t)lda + kt * 32, lda, lane);
        v16h b = load_b_frag(Wp, Nw, ct * 16, kt, lane);
        acc = wmma16(a, b, acc);
    }
    const int n = lane & 15;
    const int mb = (lane & 16) ? 8 : 0;
    const int col = ct * 16 + n;
    const float bb = bias ? bias[col] : 0.0f;
#pragma unroll
    for (int r = 0; r < 8; ++r)
        C[(rt * 16 + mb + r) * (size_t)ldc + col] = acc[r] + bb;
}

// ---------------------------------------------------------------------------
// LayerNorm: h(f16) = LN(x)*g + b. One wave per row; lane owns 8 elements.
// ---------------------------------------------------------------------------
__global__ __launch_bounds__(256) void ln_kernel(const float* __restrict__ x,
                                                 const float* __restrict__ g,
                                                 const float* __restrict__ bb,
                                                 f16* __restrict__ h) {
    const int lane = threadIdx.x & 31;
    const size_t row = (size_t)blockIdx.x * 8 + (threadIdx.x >> 5);
    const float* xr = x + row * WID_ + lane * 8;
    float4 v0 = *(const float4*)xr;
    float4 v1 = *(const float4*)(xr + 4);
    float s  = v0.x + v0.y + v0.z + v0.w + v1.x + v1.y + v1.z + v1.w;
    float sq = v0.x * v0.x + v0.y * v0.y + v0.z * v0.z + v0.w * v0.w +
               v1.x * v1.x + v1.y * v1.y + v1.z * v1.z + v1.w * v1.w;
#pragma unroll
    for (int off = 16; off > 0; off >>= 1) {
        s  += __shfl_xor(s, off, 32);
        sq += __shfl_xor(sq, off, 32);
    }
    float mean = s * (1.0f / WID_);
    float var  = sq * (1.0f / WID_) - mean * mean;
    float rstd = rsqrtf(var + 1e-5f);
    float vv[8] = {v0.x, v0.y, v0.z, v0.w, v1.x, v1.y, v1.z, v1.w};
    h8 out;
#pragma unroll
    for (int j = 0; j < 8; ++j) {
        int d = lane * 8 + j;
        out[j] = (f16)((vv[j] - mean) * rstd * g[d] + bb[d]);
    }
    *(h8*)(h + row * WID_ + lane * 8) = out;
}

// ---------------------------------------------------------------------------
// Softmax over M=32 per node, in place.
// ---------------------------------------------------------------------------
__global__ void softmax_w_kernel(float* __restrict__ w) {
    size_t t = (size_t)blockIdx.x * 256 + threadIdx.x;
    if (t >= BN_) return;
    float* p = w + t * 32;
    float v[32], mx = -1e30f;
#pragma unroll
    for (int m = 0; m < 32; ++m) { v[m] = p[m]; mx = fmaxf(mx, v[m]); }
    float sum = 0.0f;
#pragma unroll
    for (int m = 0; m < 32; ++m) { v[m] = expf(v[m] - mx); sum += v[m]; }
    float inv = 1.0f / sum;
#pragma unroll
    for (int m = 0; m < 32; ++m) p[m] = v[m] * inv;
}

__global__ void zero_kernel(float* __restrict__ p, int n) {
    int t = blockIdx.x * 256 + threadIdx.x;
    if (t < n) p[t] = 0.0f;
}

// ---------------------------------------------------------------------------
// Pooling: sacc[b,m,d] += sum_n w[b,n,m]*h[b,n,d]; wsum[b,m] += sum_n w[b,n,m]
// grid (8, B), block 256; thread owns d = tid, 32 register accumulators.
// ---------------------------------------------------------------------------
__global__ __launch_bounds__(256) void pool_kernel(const float* __restrict__ w,
                                                   const f16* __restrict__ h,
                                                   float* __restrict__ sacc,
                                                   float* __restrict__ wsum) {
    __shared__ float w_s[64 * 32];
    const int tid = threadIdx.x;
    const int b = blockIdx.y;
    const int n0c = blockIdx.x * (N_ / 8);
    float acc[32];
#pragma unroll
    for (int m = 0; m < 32; ++m) acc[m] = 0.0f;
    float wsacc = 0.0f;
    for (int s = 0; s < N_ / 8; s += 64) {
        const size_t nbase = (size_t)b * N_ + n0c + s;
        for (int i = tid; i < 64 * 32; i += 256) w_s[i] = w[nbase * 32 + i];
        __syncthreads();
        const f16* hp = h + nbase * WID_ + tid;
        for (int nl = 0; nl < 64; ++nl) {
            float hv = (float)hp[(size_t)nl * WID_];
            const float* wr = &w_s[nl * 32];
#pragma unroll
            for (int m = 0; m < 32; ++m) acc[m] += wr[m] * hv;
        }
        if (tid < 32) {
            for (int nl = 0; nl < 64; ++nl) wsacc += w_s[nl * 32 + tid];
        }
        __syncthreads();
    }
#pragma unroll
    for (int m = 0; m < 32; ++m)
        atomicAdd(&sacc[((size_t)b * 32 + m) * WID_ + tid], acc[m]);
    if (tid < 32) atomicAdd(&wsum[b * 32 + tid], wsacc);
}

__global__ void finalize_slices_kernel(const float* __restrict__ sacc,
                                       const float* __restrict__ wsum,
                                       float* __restrict__ z) {
    int t = blockIdx.x * 256 + threadIdx.x;
    if (t >= B_ * M_ * WID_) return;
    int b = t >> 13;             // / (32*256)
    int m = (t >> 8) & 31;
    z[t] = sacc[t] / fmaxf(wsum[b * 32 + m], 1e-8f);
}

// ---------------------------------------------------------------------------
// Tiny scalar GEMM: C(rows x N) = A(rows x K) @ W(K x N) + bias  (fp32)
// ---------------------------------------------------------------------------
__global__ void sgemm_kernel(const float* __restrict__ A, const float* __restrict__ W,
                             const float* __restrict__ bias, float* __restrict__ C,
                             int rows, int K, int Nw) {
    int t = blockIdx.x * 256 + threadIdx.x;
    if (t >= rows * Nw) return;
    int r = t / Nw, n = t % Nw;
    float s = bias[n];
    const float* ar = A + (size_t)r * K;
    for (int k = 0; k < K; ++k) s += ar[k] * W[(size_t)k * Nw + n];
    C[t] = s;
}

// ---------------------------------------------------------------------------
// MHA core over M=32 tokens. grid = B*H blocks, 32 threads; thread = query m.
// qkv: (B, 32, 768) [q|k|v], o: (B, 32, 256)
// ---------------------------------------------------------------------------
__global__ __launch_bounds__(32) void attn_kernel(const float* __restrict__ qkv,
                                                  float* __restrict__ o) {
    const int bh = blockIdx.x;
    const int b = bh >> 3, hd = bh & 7;
    const int m = threadIdx.x;
    float q[32], s[32], acc[32];
    const float* qp = qkv + ((size_t)(b * M_ + m)) * 768 + hd * 32;
#pragma unroll
    for (int d = 0; d < 32; ++d) q[d] = qp[d];
    float mx = -1e30f;
#pragma unroll
    for (int kk = 0; kk < 32; ++kk) {
        const float* kp = qkv + ((size_t)(b * M_ + kk)) * 768 + 256 + hd * 32;
        float t = 0.0f;
#pragma unroll
        for (int d = 0; d < 32; ++d) t += q[d] * kp[d];
        t *= 0.17677669529663687f;  // 1/sqrt(32)
        s[kk] = t;
        mx = fmaxf(mx, t);
    }
    float sum = 0.0f;
#pragma unroll
    for (int kk = 0; kk < 32; ++kk) { s[kk] = expf(s[kk] - mx); sum += s[kk]; }
    float inv = 1.0f / sum;
#pragma unroll
    for (int d = 0; d < 32; ++d) acc[d] = 0.0f;
#pragma unroll
    for (int kk = 0; kk < 32; ++kk) {
        const float* vp = qkv + ((size_t)(b * M_ + kk)) * 768 + 512 + hd * 32;
        float p = s[kk] * inv;
#pragma unroll
        for (int d = 0; d < 32; ++d) acc[d] += p * vp[d];
    }
    float* op = o + ((size_t)(b * M_ + m)) * WID_ + hd * 32;
#pragma unroll
    for (int d = 0; d < 32; ++d) op[d] = acc[d];
}

// ---------------------------------------------------------------------------
// De-slice + residual: x[b,n,d] += sum_m w[b,n,m] * sout[b,m,d]
// grid (8, B), block 256; slices_out staged once in LDS (32KB).
// ---------------------------------------------------------------------------
__global__ __launch_bounds__(256) void deslice_kernel(const float* __restrict__ w,
                                                      const float* __restrict__ sout,
                                                      float* __restrict__ x) {
    __shared__ float so_s[M_ * WID_];
    __shared__ float w_s[64 * 32];
    const int tid = threadIdx.x;
    const int b = blockIdx.y;
    const int n0c = blockIdx.x * (N_ / 8);
    for (int i = tid; i < M_ * WID_; i += 256) so_s[i] = sout[(size_t)b * M_ * WID_ + i];
    __syncthreads();
    for (int s = 0; s < N_ / 8; s += 64) {
        const size_t nbase = (size_t)b * N_ + n0c + s;
        for (int i = tid; i < 64 * 32; i += 256) w_s[i] = w[nbase * 32 + i];
        __syncthreads();
        for (int nl = 0; nl < 64; ++nl) {
            const float* wr = &w_s[nl * 32];
            float a = 0.0f;
#pragma unroll
            for (int m = 0; m < 32; ++m) a += wr[m] * so_s[m * WID_ + tid];
            x[(nbase + nl) * WID_ + tid] += a;
        }
        __syncthreads();
    }
}

// ---------------------------------------------------------------------------
// Fused FFN: x += (gelu(h2 @ W1 + b1)) @ W2 + b2, all matmuls on WMMA.
// Block: 256 threads (8 waves), 32 rows. FFN dim processed in 8 chunks of 128;
// the 32x128 f16 intermediate lives in LDS — the (B,N,1024) tensor is never
// materialized in HBM. A-tile staged via async global->LDS (ASYNCcnt path).
// ---------------------------------------------------------------------------
#define LDA_S 264   // 256 + 8 halfs pad (528B row -> bank shift of 4)
#define LDT_S 136   // 128 + 8 halfs pad

__global__ __launch_bounds__(256) void ffn_kernel(
    const f16* __restrict__ h2,   // (BN, 256) row-major f16
    const f16* __restrict__ w1p,  // panel (256,1024)
    const float* __restrict__ b1,
    const f16* __restrict__ w2p,  // panel (1024,256)
    const float* __restrict__ b2,
    float* __restrict__ x) {
    __shared__ f16 a_s[32 * LDA_S];
    __shared__ f16 t_s[32 * LDT_S];
    const int tid = threadIdx.x;
    const int wv = tid >> 5, lane = tid & 31;
    const size_t rowBase = (size_t)blockIdx.x * 32;

    // Stage the 32x256 f16 activation tile into LDS via async b128 copies.
    for (int c = tid; c < 1024; c += 256) {
        int r = c >> 5, k8 = (c & 31) << 3;
        unsigned dst = (unsigned)(uintptr_t)&a_s[r * LDA_S + k8];
        async_ld_b128(dst, &h2[(rowBase + r) * WID_ + k8]);
    }
    async_wait_all();
    __syncthreads();

    const int n = lane & 15;
    const int mb = (lane & 16) ? 8 : 0;
    const int rt = wv & 1;        // output row tile of this wave
    const int ctb = wv >> 1;      // output col tile base (+4j)
    v8f acc[4];
#pragma unroll
    for (int j = 0; j < 4; ++j)
        acc[j] = (v8f){0.f, 0.f, 0.f, 0.f, 0.f, 0.f, 0.f, 0.f};

    for (int c = 0; c < 8; ++c) {           // chunks of 128 FFN columns
        if (c + 1 < 8)
            __builtin_prefetch(&w1p[((size_t)(c + 1) * 128) * 32], 0, 1);
        // ---- ffn1: this wave computes 2 of the 16 (2x8) 16x16 t-tiles ----
#pragma unroll
        for (int tI = 0; tI < 2; ++tI) {
            const int idx = wv * 2 + tI;
            const int trt = idx >> 3, tct = idx & 7;
            v8f t = {0.f, 0.f, 0.f, 0.f, 0.f, 0.f, 0.f, 0.f};
#pragma unroll
            for (int kt = 0; kt < 8; ++kt) {   // K = 256
                v16h a = load_a_frag(&a_s[trt * 16 * LDA_S + kt * 32], LDA_S, lane);
                v16h b = load_b_frag(w1p, FFN_, c * 128 + tct * 16, kt, lane);
                t = wmma16(a, b, t);
            }
            const int cb = c * 128 + tct * 16;
            const float bb = b1[cb + n];
#pragma unroll
            for (int r = 0; r < 8; ++r) {
                float v = gelu_exact(t[r] + bb);
                t_s[(trt * 16 + mb + r) * LDT_S + tct * 16 + n] = (f16)v;
            }
        }
        __syncthreads();
        // ---- ffn2: accumulate this wave's 4 output tiles over K=128 ----
#pragma unroll
        for (int kt = 0; kt < 4; ++kt) {
            v16h a = load_a_frag(&t_s[rt * 16 * LDT_S + kt * 32], LDT_S, lane);
#pragma unroll
            for (int j = 0; j < 4; ++j) {
                const int ct = ctb + 4 * j;
                v16h b = load_b_frag(w2p, WID_, ct * 16, c * 4 + kt, lane);
                acc[j] = wmma16(a, b, acc[j]);
            }
        }
        __syncthreads();
    }
    // Epilogue: bias + residual into x.
#pragma unroll
    for (int j = 0; j < 4; ++j) {
        const int col = (ctb + 4 * j) * 16 + n;
        const float bb = b2[col];
#pragma unroll
        for (int r = 0; r < 8; ++r) {
            const size_t row = rowBase + rt * 16 + mb + r;
            x[row * WID_ + col] += acc[j][r] + bb;
        }
    }
}

// ---------------------------------------------------------------------------
// Output projection: out = x @ out_w(256x4) + out_b
// ---------------------------------------------------------------------------
__global__ void out_kernel(const float* __restrict__ x, const float* __restrict__ ow,
                           const float* __restrict__ ob, float* __restrict__ out) {
    int t = blockIdx.x * 256 + threadIdx.x;
    if (t >= BN_ * 4) return;
    int node = t >> 2, oc = t & 3;
    const float* xr = x + (size_t)node * WID_;
    float s = ob[oc];
    for (int k = 0; k < WID_; ++k) s += xr[k] * ow[k * 4 + oc];
    out[t] = s;
}

// ---------------------------------------------------------------------------
// Host launcher
// ---------------------------------------------------------------------------
extern "C" void kernel_launch(void* const* d_in, const int* in_sizes, int n_in,
                              void* d_out, int out_size, void* d_ws, size_t ws_size,
                              hipStream_t stream) {
    (void)in_sizes; (void)n_in; (void)out_size; (void)ws_size;
    const float* features   = (const float*)d_in[0];
    const float* coords     = (const float*)d_in[1];
    const float* t_norm     = (const float*)d_in[2];
    const float* freq       = (const float*)d_in[3];
    const float* omega      = (const float*)d_in[4];
    const float* in_w       = (const float*)d_in[5];
    const float* in_b       = (const float*)d_in[6];
    const float* ln1_g      = (const float*)d_in[7];
    const float* ln1_b      = (const float*)d_in[8];
    const float* slice_w    = (const float*)d_in[9];
    const float* slice_b    = (const float*)d_in[10];
    const float* qkv_w      = (const float*)d_in[11];
    const float* qkv_b      = (const float*)d_in[12];
    const float* attn_out_w = (const float*)d_in[13];
    const float* attn_out_b = (const float*)d_in[14];
    const float* ln2_g      = (const float*)d_in[15];
    const float* ln2_b      = (const float*)d_in[16];
    const float* ffn1_w     = (const float*)d_in[17];
    const float* ffn1_b     = (const float*)d_in[18];
    const float* ffn2_w     = (const float*)d_in[19];
    const float* ffn2_b     = (const float*)d_in[20];
    const float* out_w      = (const float*)d_in[21];
    const float* out_b      = (const float*)d_in[22];

    // --- workspace carve (256B aligned) ---
    char* base = (char*)d_ws;
    size_t off = 0;
    auto take = [&](size_t bytes) -> void* {
        void* p = base + off;
        off = (off + bytes + 255) & ~(size_t)255;
        return p;
    };
    float* x       = (float*)take((size_t)BN_ * WID_ * 4);          // 134 MB
    f16*   hbuf    = (f16*)  take((size_t)BN_ * WID_ * 2);          //  67 MB
    float* wbuf    = (float*)take((size_t)BN_ * M_ * 4);            //  17 MB
    f16*   enc     = (f16*)  take((size_t)BN_ * 32 * 2);            // 8.4 MB
    f16*   inw_p   = (f16*)  take((size_t)32 * WID_ * 2);
    f16*   slw_p   = (f16*)  take((size_t)DEPTH_ * WID_ * M_ * 2);
    f16*   f1_p    = (f16*)  take((size_t)DEPTH_ * WID_ * FFN_ * 2);
    f16*   f2_p    = (f16*)  take((size_t)DEPTH_ * FFN_ * WID_ * 2);
    float* sacc    = (float*)take((size_t)(B_ * M_ * WID_ + B_ * M_) * 4); // +wsum
    float* wsum    = sacc + (size_t)B_ * M_ * WID_;
    float* z       = (float*)take((size_t)B_ * M_ * WID_ * 4);
    float* qkvb    = (float*)take((size_t)B_ * M_ * 768 * 4);
    float* obuf    = (float*)take((size_t)B_ * M_ * WID_ * 4);
    float* soutb   = (float*)take((size_t)B_ * M_ * WID_ * 4);

    // --- pack weights to f16 panel layout (deterministic, every call) ---
    pack_w_kernel<<<dim3((32 * 256 + 255) / 256, 1), 256, 0, stream>>>(
        in_w, inw_p, 28, 32, 256);
    pack_w_kernel<<<dim3((256 * 32 + 255) / 256, DEPTH_), 256, 0, stream>>>(
        slice_w, slw_p, 256, 256, 32);
    pack_w_kernel<<<dim3((256 * 1024 + 255) / 256, DEPTH_), 256, 0, stream>>>(
        ffn1_w, f1_p, 256, 256, 1024);
    pack_w_kernel<<<dim3((1024 * 256 + 255) / 256, DEPTH_), 256, 0, stream>>>(
        ffn2_w, f2_p, 1024, 1024, 256);

    // --- encoding + input projection (WMMA, K padded to 32) ---
    encode_kernel<<<BN_ / 256, 256, 0, stream>>>(features, coords, t_norm, freq,
                                                 omega, enc);
    wmma_gemm_kernel<<<dim3(BN_ / 16, WID_ / 16), 32, 0, stream>>>(
        enc, 32, inw_p, in_b, x, WID_, 32, WID_);

    for (int i = 0; i < DEPTH_; ++i) {
        // h = LN1(x)
        ln_kernel<<<BN_ / 8, 256, 0, stream>>>(x, ln1_g + i * WID_,
                                               ln1_b + i * WID_, hbuf);
        // slice logits -> wbuf (WMMA), softmax over M
        wmma_gemm_kernel<<<dim3(BN_ / 16, M_ / 16), 32, 0, stream>>>(
            hbuf, WID_, slw_p + (size_t)i * WID_ * M_, slice_b + i * M_,
            wbuf, M_, WID_, M_);
        softmax_w_kernel<<<BN_ / 256, 256, 0, stream>>>(wbuf);
        // pooled slices
        zero_kernel<<<(B_ * M_ * WID_ + B_ * M_ + 255) / 256, 256, 0, stream>>>(
            sacc, B_ * M_ * WID_ + B_ * M_);
        pool_kernel<<<dim3(8, B_), 256, 0, stream>>>(wbuf, hbuf, sacc, wsum);
        finalize_slices_kernel<<<(B_ * M_ * WID_ + 255) / 256, 256, 0, stream>>>(
            sacc, wsum, z);
        // MHA over 32 tokens (tiny; scalar fp32)
        sgemm_kernel<<<(B_ * M_ * 768 + 255) / 256, 256, 0, stream>>>(
            z, qkv_w + (size_t)i * WID_ * 768, qkv_b + i * 768, qkvb,
            B_ * M_, WID_, 768);
        attn_kernel<<<B_ * H_, 32, 0, stream>>>(qkvb, obuf);
        sgemm_kernel<<<(B_ * M_ * WID_ + 255) / 256, 256, 0, stream>>>(
            obuf, attn_out_w + (size_t)i * WID_ * WID_, attn_out_b + i * WID_,
            soutb, B_ * M_, WID_, WID_);
        // de-slice + residual into x
        deslice_kernel<<<dim3(8, B_), 256, 0, stream>>>(wbuf, soutb, x);
        // h2 = LN2(x); fused WMMA FFN + residual
        ln_kernel<<<BN_ / 8, 256, 0, stream>>>(x, ln2_g + i * WID_,
                                               ln2_b + i * WID_, hbuf);
        ffn_kernel<<<BN_ / 32, 256, 0, stream>>>(
            hbuf, f1_p + (size_t)i * WID_ * FFN_, ffn1_b + i * FFN_,
            f2_p + (size_t)i * FFN_ * WID_, ffn2_b + i * WID_, x);
    }
    out_kernel<<<(BN_ * 4 + 255) / 256, 256, 0, stream>>>(x, out_w, out_b,
                                                          (float*)d_out);
}
